// HarmonicGPT_30571577213599
// MI455X (gfx1250) — matrix operand order repaired and checked by reading, not compiled, not run on hardware
//
#include <hip/hip_runtime.h>
#include <hip/hip_bf16.h>
#include <math.h>

// ---------------- model dims (compile-time) ----------------
constexpr int gB  = 4;
constexpr int gT  = 1024;
constexpr int gC  = 768;
constexpr int gH  = 12;
constexpr int gL  = 6;
constexpr int gV  = 32000;
constexpr int gHD = gC / gH;      // 64
constexpr int gM  = gB * gT;      // 4096 rows

typedef __attribute__((ext_vector_type(16))) __bf16 v16bf;
typedef __attribute__((ext_vector_type(8)))  __bf16 v8bf;
typedef __attribute__((ext_vector_type(8)))  float  v8f;

static __device__ __forceinline__ v16bf mk16(v8bf lo, v8bf hi) {
  v16bf r;
#pragma unroll
  for (int i = 0; i < 8; ++i) { r[i] = lo[i]; r[i + 8] = hi[i]; }
  return r;
}
static __device__ __forceinline__ v8f zero8() {
  v8f z;
#pragma unroll
  for (int i = 0; i < 8; ++i) z[i] = 0.0f;
  return z;
}

// ---------------- weight convert + transpose: f32 [K,N] -> bf16 [N,K] ----------------
__global__ void cvt_transpose_kernel(const float* __restrict__ in, __bf16* __restrict__ out,
                                     int K, int N) {
  size_t i = (size_t)blockIdx.x * 256 + threadIdx.x;
  if (i >= (size_t)K * N) return;
  int k = (int)(i / N), n = (int)(i - (size_t)k * N);
  out[(size_t)n * K + k] = (__bf16)in[i];
}

// ---------------- token + position embedding ----------------
__global__ void embed_kernel(const int* __restrict__ idx, const float* __restrict__ wte,
                             const float* __restrict__ wpe, float* __restrict__ x) {
  int bt = blockIdx.x;
  int t  = bt % gT;
  int tok = idx[bt];
  const float* te = wte + (size_t)tok * gC;
  const float* pe = wpe + (size_t)t * gC;
  float* xo = x + (size_t)bt * gC;
  for (int c = threadIdx.x; c < gC; c += 256) xo[c] = te[c] + pe[c];
}

// ---------------- layernorm (fp32 in, bf16 out) ----------------
__global__ __launch_bounds__(256) void layernorm_kernel(const float* __restrict__ x,
                                                        const float* __restrict__ w,
                                                        const float* __restrict__ b,
                                                        __bf16* __restrict__ out) {
  __shared__ float red[256];
  const int row = blockIdx.x, tid = threadIdx.x;
  const float* xr = x + (size_t)row * gC;
  float s = 0.0f;
  for (int c = tid; c < gC; c += 256) s += xr[c];
  red[tid] = s; __syncthreads();
  for (int st = 128; st > 0; st >>= 1) { if (tid < st) red[tid] += red[tid + st]; __syncthreads(); }
  const float mu = red[0] * (1.0f / gC);
  __syncthreads();
  float v = 0.0f;
  for (int c = tid; c < gC; c += 256) { float d = xr[c] - mu; v += d * d; }
  red[tid] = v; __syncthreads();
  for (int st = 128; st > 0; st >>= 1) { if (tid < st) red[tid] += red[tid + st]; __syncthreads(); }
  const float rstd = rsqrtf(red[0] * (1.0f / gC) + 1e-5f);
  __bf16* orow = out + (size_t)row * gC;
  for (int c = tid; c < gC; c += 256)
    orow[c] = (__bf16)((xr[c] - mu) * rstd * w[c] + b[c]);
}

// ---------------- split fused qkv (f32) into q,k (bf16 [BH,T,HD]) and v^T (bf16 [BH,HD,T]) ----
__global__ void split_qkv_kernel(const float* __restrict__ qkv, __bf16* __restrict__ qf,
                                 __bf16* __restrict__ kf, __bf16* __restrict__ vT) {
  int i = blockIdx.x * 256 + threadIdx.x;     // over gM*gC
  if (i >= gM * gC) return;
  int bt = i / gC, c = i - bt * gC;
  int b = bt / gT, t = bt - b * gT;
  int h = c / gHD, hd = c - h * gHD;
  int bh = b * gH + h;
  const float* row = qkv + (size_t)bt * (3 * gC);
  qf[((size_t)bh * gT + t) * gHD + hd] = (__bf16)row[c];
  kf[((size_t)bh * gT + t) * gHD + hd] = (__bf16)row[gC + c];
  vT[((size_t)bh * gHD + hd) * gT + t] = (__bf16)row[2 * gC + c];
}

// ---------------- WMMA GEMM: out[M,N] = A_bf16[M,K] @ (BT_bf16[N,K])^T (+bias)(+gelu)(+res) ----
// block = 256 threads = 8 waves; block tile 64(M) x 256(N) x 32(K); wave tile 32x64
// (2x4 accumulators, 8 WMMAs per K-step per wave; each A frag feeds 4 WMMAs)
template <bool HAS_BIAS, bool HAS_RES, bool DO_GELU, bool OUT_F32>
__global__ __launch_bounds__(256) void gemm_bf16_wmma(const __bf16* __restrict__ A,
                                                      const __bf16* __restrict__ BT,
                                                      const float* __restrict__ bias,
                                                      const float* __restrict__ res,
                                                      float* __restrict__ outF,
                                                      __bf16* __restrict__ outB,
                                                      int M, int N, int K) {
  constexpr int BM = 64, BN = 256, BK = 32;
  __shared__ __align__(16) __bf16 sA[BM * BK];   //  4 KB
  __shared__ __align__(16) __bf16 sB[BN * BK];   // 16 KB
  const int tid = threadIdx.x;
  const int wave = tid >> 5, lane = tid & 31;
  const int bm = blockIdx.y * BM, bn = blockIdx.x * BN;
  const int wm = (wave >> 2) * 32, wn = (wave & 3) * 64;
  const int lr = lane & 15, lh = lane >> 4;

  v8f acc[2][4];
#pragma unroll
  for (int i = 0; i < 2; ++i)
#pragma unroll
    for (int j = 0; j < 4; ++j) acc[i][j] = zero8();

  for (int k0 = 0; k0 < K; k0 += BK) {
    __syncthreads();
    {  // stage A tile: 64 rows x 32 halves, one b128 per thread
      int r = tid >> 2, c = (tid & 3) * 8;
      *(uint4*)&sA[r * BK + c] = *(const uint4*)&A[(size_t)(bm + r) * K + (k0 + c)];
      if (k0 + BK < K) __builtin_prefetch(&A[(size_t)(bm + r) * K + (k0 + BK + c)], 0, 0);
    }
#pragma unroll
    for (int i = 0; i < 4; ++i) {  // stage B^T tile: 256 rows(n) x 32 halves(k)
      int id = tid + 256 * i;
      int r = id >> 2, c = (id & 3) * 8;
      *(uint4*)&sB[r * BK + c] = *(const uint4*)&BT[(size_t)(bn + r) * K + (k0 + c)];
      if (k0 + BK < K) __builtin_prefetch(&BT[(size_t)(bn + r) * K + (k0 + BK + c)], 0, 0);
    }
    __syncthreads();

    v16bf af[2], bfr[4];
#pragma unroll
    for (int i = 0; i < 2; ++i) {   // A fragment: lane=row, halves cover K per ISA layout
      const __bf16* p = &sA[(wm + i * 16 + lr) * BK];
      af[i] = mk16(*(const v8bf*)&p[lh * 8], *(const v8bf*)&p[lh * 8 + 16]);
    }
#pragma unroll
    for (int j = 0; j < 4; ++j) {   // B fragment: lane=col, halves cover K per ISA layout
      const __bf16* p = &sB[(wn + j * 16 + lr) * BK];
      bfr[j] = mk16(*(const v8bf*)&p[lh * 8], *(const v8bf*)&p[lh * 8 + 16]);
    }
#pragma unroll
    for (int i = 0; i < 2; ++i)
#pragma unroll
      for (int j = 0; j < 4; ++j)
        acc[i][j] = __builtin_amdgcn_wmma_f32_16x16x32_bf16(
            false, af[i], false, bfr[j], (short)0, acc[i][j], false, false);
  }

#pragma unroll
  for (int i = 0; i < 2; ++i)
#pragma unroll
    for (int j = 0; j < 4; ++j)
#pragma unroll
      for (int r = 0; r < 8; ++r) {
        int row = bm + wm + i * 16 + r + lh * 8;
        int col = bn + wn + j * 16 + lr;
        float v = acc[i][j][r];
        if (HAS_BIAS) v += bias[col];
        if (DO_GELU)  v = 0.5f * v * (1.0f + erff(v * 0.70710678118654752f));
        size_t o = (size_t)row * N + col;
        if (HAS_RES)  v += res[o];
        if (OUT_F32) outF[o] = v; else outB[o] = (__bf16)v;
      }
}

// ---------------- flash attention: 1 wave per 16-row Q tile, online softmax ----------------
__global__ __launch_bounds__(128) void attention_wmma(const __bf16* __restrict__ qf,
                                                      const __bf16* __restrict__ kf,
                                                      const __bf16* __restrict__ vT,
                                                      __bf16* __restrict__ ybf) {
  constexpr float scale = 0.125f;  // 1/sqrt(64)
  __shared__ __align__(16) __bf16 sP[4][16 * 32];  // per-wave P tile (A-format staging)
  const int wave = threadIdx.x >> 5, lane = threadIdx.x & 31;
  const int lr = lane & 15, lh = lane >> 4;
  const int bh = blockIdx.y;
  const int b = bh / gH, h = bh - b * gH;
  const int q0 = (blockIdx.x * 4 + wave) * 16;
  const size_t base = (size_t)bh * gT * gHD;

  // Q fragments (A layout): row = q0+lr, hd chunks 0..31 / 32..63
  v16bf aQ0, aQ1;
  {
    const __bf16* p = qf + base + (size_t)(q0 + lr) * gHD;
    aQ0 = mk16(*(const v8bf*)&p[lh * 8],      *(const v8bf*)&p[lh * 8 + 16]);
    aQ1 = mk16(*(const v8bf*)&p[32 + lh * 8], *(const v8bf*)&p[32 + lh * 8 + 16]);
  }

  v8f o[4];
  float m_r[8], l_r[8];
#pragma unroll
  for (int j = 0; j < 4; ++j) o[j] = zero8();
#pragma unroll
  for (int r = 0; r < 8; ++r) { m_r[r] = -1e30f; l_r[r] = 0.0f; }
  __bf16* pw = sP[wave];

  const int nb = (q0 + 47) >> 5;  // 32-key blocks covering keys <= q0+15
  for (int kb = 0; kb < nb; ++kb) {
    const int k0 = kb * 32;
    // S = Q @ K^T for 16x32 keys (two 16x16 tiles, chained WMMAs over HD=64)
    v8f s0, s1;
    {
      const __bf16* pk0 = kf + base + (size_t)(k0 + lr) * gHD + lh * 8;
      const __bf16* pk1 = kf + base + (size_t)(k0 + 16 + lr) * gHD + lh * 8;
      v16bf b00 = mk16(*(const v8bf*)&pk0[0],  *(const v8bf*)&pk0[16]);
      v16bf b01 = mk16(*(const v8bf*)&pk0[32], *(const v8bf*)&pk0[48]);
      v16bf b10 = mk16(*(const v8bf*)&pk1[0],  *(const v8bf*)&pk1[16]);
      v16bf b11 = mk16(*(const v8bf*)&pk1[32], *(const v8bf*)&pk1[48]);
      s0 = __builtin_amdgcn_wmma_f32_16x16x32_bf16(false, aQ0, false, b00, (short)0, zero8(), false, false);
      s0 = __builtin_amdgcn_wmma_f32_16x16x32_bf16(false, aQ1, false, b01, (short)0, s0,      false, false);
      s1 = __builtin_amdgcn_wmma_f32_16x16x32_bf16(false, aQ0, false, b10, (short)0, zero8(), false, false);
      s1 = __builtin_amdgcn_wmma_f32_16x16x32_bf16(false, aQ1, false, b11, (short)0, s1,      false, false);
    }
    // online softmax; each accumulator VGPR r holds row q0+r+8*lh across its 16-lane group
#pragma unroll
    for (int r = 0; r < 8; ++r) {
      const int qg = q0 + r + lh * 8;
      float x0 = s0[r] * scale; if (k0 + lr      > qg) x0 = -1e30f;
      float x1 = s1[r] * scale; if (k0 + 16 + lr > qg) x1 = -1e30f;
      float mx = fmaxf(x0, x1);
#pragma unroll
      for (int off = 1; off < 16; off <<= 1) mx = fmaxf(mx, __shfl_xor(mx, off, 32));
      const float mn = fmaxf(m_r[r], mx);
      const float p0 = __expf(x0 - mn), p1 = __expf(x1 - mn);
      float rs = p0 + p1;
#pragma unroll
      for (int off = 1; off < 16; off <<= 1) rs += __shfl_xor(rs, off, 32);
      const float alpha = __expf(m_r[r] - mn);
      l_r[r] = l_r[r] * alpha + rs;
      m_r[r] = mn;
#pragma unroll
      for (int j = 0; j < 4; ++j) o[j][r] *= alpha;
      const int m = r + lh * 8;
      pw[m * 32 + lr]      = (__bf16)p0;   // C-layout -> row-major P tile in LDS
      pw[m * 32 + 16 + lr] = (__bf16)p1;
    }
    asm volatile("s_wait_dscnt 0" ::: "memory");  // LDS RAW fence (same wave, in-order pipe)
    v16bf aP;
    {
      const __bf16* p = &pw[lr * 32];
      aP = mk16(*(const v8bf*)&p[lh * 8], *(const v8bf*)&p[lh * 8 + 16]);
    }
#pragma unroll
    for (int j = 0; j < 4; ++j) {  // O += P @ V ; V stored transposed [BH,HD,T]
      const __bf16* pv = vT + (size_t)(bh * gHD + j * 16 + lr) * gT + k0 + lh * 8;
      v16bf bV = mk16(*(const v8bf*)&pv[0], *(const v8bf*)&pv[16]);
      o[j] = __builtin_amdgcn_wmma_f32_16x16x32_bf16(false, aP, false, bV, (short)0, o[j], false, false);
    }
  }

  // write y (bf16, [B,T,C] head-interleaved) = O / l
#pragma unroll
  for (int r = 0; r < 8; ++r) {
    const float inv = 1.0f / l_r[r];
    const int trow = q0 + r + lh * 8;
    const size_t ob = ((size_t)(b * gT + trow)) * gC + h * gHD + lr;
#pragma unroll
    for (int j = 0; j < 4; ++j) ybf[ob + j * 16] = (__bf16)(o[j][r] * inv);
  }
}

// ============================== host launch ==============================
extern "C" void kernel_launch(void* const* d_in, const int* in_sizes, int n_in,
                              void* d_out, int out_size, void* d_ws, size_t ws_size,
                              hipStream_t stream) {
  (void)in_sizes; (void)n_in; (void)out_size; (void)ws_size;
  const int*   idx    = (const int*)d_in[0];
  const float* wte    = (const float*)d_in[1];
  const float* wpe    = (const float*)d_in[2];
  const float* ln1_w  = (const float*)d_in[3];
  const float* ln1_b  = (const float*)d_in[4];
  const float* attn_w = (const float*)d_in[5];
  const float* attn_b = (const float*)d_in[6];
  const float* proj_w = (const float*)d_in[7];
  const float* proj_b = (const float*)d_in[8];
  const float* ln2_w  = (const float*)d_in[9];
  const float* ln2_b  = (const float*)d_in[10];
  const float* fc_w   = (const float*)d_in[11];
  const float* fc_b   = (const float*)d_in[12];
  const float* mlp_w  = (const float*)d_in[13];
  const float* mlp_b  = (const float*)d_in[14];
  const float* lnf_w  = (const float*)d_in[15];
  const float* lnf_b  = (const float*)d_in[16];
  const float* head_w = (const float*)d_in[17];
  float* logits = (float*)d_out;

  // workspace layout (~156 MB), 256B aligned
  char* wsp = (char*)d_ws;
  auto alloc = [&](size_t bytes) -> char* {
    char* p = wsp;
    wsp += (bytes + 255) & ~(size_t)255;
    return p;
  };
  float*  x     = (float*) alloc((size_t)gM * gC * 4);
  __bf16* lnb   = (__bf16*)alloc((size_t)gM * gC * 2);
  float*  qkv   = (float*) alloc((size_t)gM * 3 * gC * 4);
  __bf16* qfb   = (__bf16*)alloc((size_t)gM * gC * 2);
  __bf16* kfb   = (__bf16*)alloc((size_t)gM * gC * 2);
  __bf16* vTb   = (__bf16*)alloc((size_t)gM * gC * 2);
  __bf16* ybf   = (__bf16*)alloc((size_t)gM * gC * 2);
  __bf16* hbf   = (__bf16*)alloc((size_t)gM * 4 * gC * 2);
  __bf16* wT    = (__bf16*)alloc((size_t)gV * gC * 2);   // reusable W^T bf16 scratch (max = head)

  auto cvtT = [&](const float* in, __bf16* out, int K, int N) {
    size_t tot = (size_t)K * N;
    cvt_transpose_kernel<<<(unsigned)((tot + 255) / 256), 256, 0, stream>>>(in, out, K, N);
  };

  embed_kernel<<<gM, 256, 0, stream>>>(idx, wte, wpe, x);

  const dim3 gGemmQKV(3 * gC / 256, gM / 64);   // (9, 64)
  const dim3 gGemmC  (gC / 256,     gM / 64);   // (3, 64)
  const dim3 gGemmFC (4 * gC / 256, gM / 64);   // (12, 64)
  const dim3 gGemmHD (gV / 256,     gM / 64);   // (125, 64)
  const dim3 gAttn   (gT / 16 / 4,  gB * gH);   // (16, 48)

  for (int l = 0; l < gL; ++l) {
    // ---- attention block ----
    layernorm_kernel<<<gM, 256, 0, stream>>>(x, ln1_w + l * gC, ln1_b + l * gC, lnb);
    cvtT(attn_w + (size_t)l * gC * 3 * gC, wT, gC, 3 * gC);
    gemm_bf16_wmma<true, false, false, true><<<gGemmQKV, 256, 0, stream>>>(
        lnb, wT, attn_b + l * 3 * gC, nullptr, qkv, nullptr, gM, 3 * gC, gC);
    split_qkv_kernel<<<(gM * gC + 255) / 256, 256, 0, stream>>>(qkv, qfb, kfb, vTb);
    attention_wmma<<<gAttn, 128, 0, stream>>>(qfb, kfb, vTb, ybf);
    cvtT(proj_w + (size_t)l * gC * gC, wT, gC, gC);
    gemm_bf16_wmma<true, true, false, true><<<gGemmC, 256, 0, stream>>>(
        ybf, wT, proj_b + l * gC, x, x, nullptr, gM, gC, gC);
    // ---- mlp block ----
    layernorm_kernel<<<gM, 256, 0, stream>>>(x, ln2_w + l * gC, ln2_b + l * gC, lnb);
    cvtT(fc_w + (size_t)l * gC * 4 * gC, wT, gC, 4 * gC);
    gemm_bf16_wmma<true, false, true, false><<<gGemmFC, 256, 0, stream>>>(
        lnb, wT, fc_b + l * 4 * gC, nullptr, nullptr, hbf, gM, 4 * gC, gC);
    cvtT(mlp_w + (size_t)l * 4 * gC * gC, wT, 4 * gC, gC);
    gemm_bf16_wmma<true, true, false, true><<<gGemmC, 256, 0, stream>>>(
        hbf, wT, mlp_b + l * gC, x, x, nullptr, gM, gC, 4 * gC);
  }

  layernorm_kernel<<<gM, 256, 0, stream>>>(x, lnf_w, lnf_b, lnb);
  cvtT(head_w, wT, gC, gV);
  gemm_bf16_wmma<false, false, false, true><<<gGemmHD, 256, 0, stream>>>(
      lnb, wT, nullptr, nullptr, logits, nullptr, gM, gV, gC);
}
